// Basic_Block_20779051778118
// MI455X (gfx1250) — compile-verified, matrix operand order checked by reading, not run on hardware
//
#include <hip/hip_runtime.h>
#include <hip/hip_bf16.h>
#include <math.h>

// ---------------- problem constants ----------------
#define HDIM   512
#define NHEAD  8
#define DHEAD  64
#define SLEN   2048
#define BATCH  4
#define MROWS  (BATCH * SLEN)      // 8192
#define CHUNK  256                 // attention key chunk held in LDS
#define NCHUNK (SLEN / CHUNK)      // 8

// GEMM tiling
#define BN      64                 // N columns per block
#define KH      256                // K staged per phase (2 phases of 512)
#define BPAD    8                  // bf16 pad per LDS row (bank decorrelation)
#define BSTRIDE (KH + BPAD)        // 264 bf16 per LDS row

typedef __bf16 bf16_t;
typedef __attribute__((ext_vector_type(16))) bf16_t v16bf;
typedef __attribute__((ext_vector_type(8)))  bf16_t v8bf;
typedef __attribute__((ext_vector_type(8)))  float  v8f;

// ---------------- helpers ----------------
__device__ inline v8f vzero8() {
  v8f z;
#pragma unroll
  for (int i = 0; i < 8; ++i) z[i] = 0.0f;
  return z;
}

// Load one 16-bit WMMA operand fragment for this lane: 8 bf16 at p[0..7] and
// 8 bf16 at p[16..23] (caller bakes in the half-wave K offset and row select).
__device__ inline v16bf load_frag(const bf16_t* p) {
  v8bf lo = *(const v8bf*)(p);
  v8bf hi = *(const v8bf*)(p + 16);
  return __builtin_shufflevector(lo, hi, 0, 1, 2, 3, 4, 5, 6, 7,
                                 8, 9, 10, 11, 12, 13, 14, 15);
}

__device__ inline v8f wmma_bf16(v16bf a, v16bf b, v8f c) {
  // (neg_a, A, neg_b, B, c_mod, C, reuse_a, reuse_b)
  return __builtin_amdgcn_wmma_f32_16x16x32_bf16(false, a, false, b,
                                                 (short)0, c, false, false);
}

__device__ inline float gelu_erf(float x) {
  return 0.5f * x * (1.0f + erff(x * 0.70710678118654752f));
}

// ---------------- prep kernels ----------------
// W fp32 [K=512][N=512] row-major -> Wt bf16 [N][K].
__global__ __launch_bounds__(256) void transpose_w_kernel(const float* __restrict__ W,
                                                          bf16_t* __restrict__ Wt) {
  int idx = blockIdx.x * 256 + threadIdx.x;   // 0 .. 512*512-1
  int k = idx >> 9;
  int n = idx & 511;
  Wt[(size_t)n * HDIM + k] = (bf16_t)W[idx];
}

__global__ __launch_bounds__(256) void cvt_f32_bf16_kernel(const float* __restrict__ src,
                                                           bf16_t* __restrict__ dst, int n) {
  int i = blockIdx.x * 256 + threadIdx.x;
  if (i < n) dst[i] = (bf16_t)src[i];
}

// vb bf16 [B,S,H] -> vt bf16 [B,NH,DH,S]  (keys contiguous per head-dim row)
__global__ __launch_bounds__(256) void transpose_v_kernel(const bf16_t* __restrict__ vb,
                                                          bf16_t* __restrict__ vt) {
  int idx = blockIdx.x * 256 + threadIdx.x;   // 0 .. B*S*H-1
  int c = idx & (HDIM - 1);
  int s = (idx >> 9) & (SLEN - 1);
  int b = idx >> 20;                          // S*H = 2^20
  int h = c >> 6;
  int d = c & (DHEAD - 1);
  vt[(((size_t)(b * NHEAD + h) * DHEAD) + d) * SLEN + s] = vb[idx];
}

// ---------------- WMMA GEMM with async-LDS weight staging ----------------
// C[M x 512] = A_bf16[M x 512] * Wt_bf16[N][K].
// Block = 8 waves = 128(M) x 64(N) region. The block's 64x512 bf16 weight
// strip is staged into LDS in two K-halves with GLOBAL_LOAD_ASYNC_TO_LDS_B128
// (ASYNCcnt), then each wave runs 4 WMMAs per A-fragment from LDS B-fragments.
__global__ __launch_bounds__(256) void gemm_bf16_kernel(
    const bf16_t* __restrict__ A, const bf16_t* __restrict__ Bw,
    const float* __restrict__ bias, const float* __restrict__ resid,
    float* __restrict__ outf, bf16_t* __restrict__ outb,
    int M, int do_gelu) {
  __shared__ bf16_t Bs[64 * BSTRIDE];         // 33,792 bytes

  int t    = threadIdx.x;
  int wave = t >> 5;
  int lane = t & 31;
  int half = lane >> 4;
  int r    = lane & 15;

  int bn = blockIdx.x & 7;                    // 512 / BN = 8 strips in N
  int bm = blockIdx.x >> 3;
  int m0 = bm * 128 + wave * 16;
  if (m0 >= M) return;

  // LDS byte offset of Bs: flat shared address low 32 bits == LDS offset.
  unsigned ldsbase = (unsigned)(uintptr_t)(&Bs[0]);

  const bf16_t* ap_base = A + (size_t)(m0 + r) * HDIM + half * 8;

  v8f acc[4];
#pragma unroll
  for (int ns = 0; ns < 4; ++ns) acc[ns] = vzero8();

  for (int ph = 0; ph < 2; ++ph) {
    __syncthreads();                          // previous phase consumers done
    // stage 64 rows x 256 bf16 (512B = 32 x 16B chunks per row) of Bw
#pragma unroll
    for (int it = 0; it < 8; ++it) {
      int chunk = it * 256 + t;               // 0 .. 2047
      int row = chunk >> 5;
      int cc  = chunk & 31;
      const bf16_t* gp =
          Bw + (size_t)(bn * BN + row) * HDIM + ph * KH + cc * 8;
      unsigned lo = ldsbase + (unsigned)(row * BSTRIDE * 2 + cc * 16);
      asm volatile("global_load_async_to_lds_b128 %0, %1, off"
                   :: "v"(lo), "v"(gp) : "memory");
    }
    asm volatile("s_wait_asynccnt 0x0" ::: "memory");
    __syncthreads();                          // all waves' stages visible

#pragma unroll 2
    for (int k0 = 0; k0 < KH; k0 += 32) {
      __builtin_prefetch(ap_base + ph * KH + k0 + 64, 0, 0);
      v16bf a = load_frag(ap_base + ph * KH + k0);
#pragma unroll
      for (int ns = 0; ns < 4; ++ns) {
        const bf16_t* bp = &Bs[(ns * 16 + r) * BSTRIDE + k0 + half * 8];
        v16bf b = load_frag(bp);
        acc[ns] = wmma_bf16(a, b, acc[ns]);
      }
    }
  }

  // epilogue
#pragma unroll
  for (int ns = 0; ns < 4; ++ns) {
    int col = bn * BN + ns * 16 + r;
    float bi = bias ? bias[col] : 0.0f;
#pragma unroll
    for (int i = 0; i < 8; ++i) {
      int row = m0 + i + half * 8;
      float v = acc[ns][i] + bi;
      if (do_gelu) v = gelu_erf(v);
      size_t idx = (size_t)row * HDIM + col;
      if (resid) v += resid[idx];
      if (outf) outf[idx] = v;
      if (outb) outb[idx] = (bf16_t)v;
    }
  }
}

// ---------------- attention (flash-style, WMMA) ----------------
// One wave per (b, h, 16-query block). Online softmax over 256-key chunks
// staged in LDS as fp32; scores and P*V both via v_wmma_f32_16x16x32_bf16.
__global__ __launch_bounds__(64) void attn_kernel(
    const bf16_t* __restrict__ qb, const bf16_t* __restrict__ kb,
    const bf16_t* __restrict__ vt, bf16_t* __restrict__ ob) {
  __shared__ float sm[2][16 * CHUNK + 16];    // per-wave: scores + row factors

  int w    = threadIdx.x >> 5;
  int lane = threadIdx.x & 31;
  int half = lane >> 4;
  int r    = lane & 15;

  int wid  = blockIdx.x * 2 + w;              // 0 .. B*NH*(S/16)-1 = 4095
  int qblk = wid & 127;                       // S/16 = 128
  int h    = (wid >> 7) & (NHEAD - 1);
  int b    = wid >> 10;
  int q0   = qblk * 16;

  float* Ssc = &sm[w][0];
  float* fac = &sm[w][16 * CHUNK];

  // Q fragments are invariant over the key loop: load once.
  const bf16_t* qptr = qb + ((size_t)(b * SLEN + q0 + r)) * HDIM + h * DHEAD;
  v16bf aq0 = load_frag(qptr + half * 8);         // d = 0..31
  v16bf aq1 = load_frag(qptr + 32 + half * 8);    // d = 32..63

  float m_r = -1e30f, l_r = 0.0f;                 // rows owned by lanes 0..15
  v8f acc[4];
#pragma unroll
  for (int dt = 0; dt < 4; ++dt) acc[dt] = vzero8();

  for (int ck = 0; ck < NCHUNK; ++ck) {
    int kc = ck * CHUNK;

    // ---- scores for 16 key tiles -> LDS (scaled by 1/sqrt(DH)) ----
    for (int kt = 0; kt < 16; ++kt) {
      const bf16_t* kp =
          kb + ((size_t)(b * SLEN + kc + kt * 16 + r)) * HDIM + h * DHEAD;
      v16bf bk0 = load_frag(kp + half * 8);
      v16bf bk1 = load_frag(kp + 32 + half * 8);
      v8f c = vzero8();
      c = wmma_bf16(aq0, bk0, c);
      c = wmma_bf16(aq1, bk1, c);
#pragma unroll
      for (int i = 0; i < 8; ++i)
        Ssc[(i + half * 8) * CHUNK + kt * 16 + r] = c[i] * 0.125f;
    }

    // ---- online softmax stats, lanes 0..15 each own one query row ----
    if (lane < 16) {
      float* row = Ssc + lane * CHUNK;
      float cm = -1e30f;
      for (int j = 0; j < CHUNK; ++j) cm = fmaxf(cm, row[j]);
      float nm = fmaxf(m_r, cm);
      float f  = __expf(m_r - nm);
      float ss = 0.0f;
      for (int j = 0; j < CHUNK; ++j) {
        float p = __expf(row[j] - nm);
        row[j] = p;
        ss += p;
      }
      l_r = l_r * f + ss;
      m_r = nm;
      fac[lane] = f;
    }

    // ---- rescale accumulators by per-row correction factor ----
    float fv[8];
#pragma unroll
    for (int i = 0; i < 8; ++i) fv[i] = fac[i + half * 8];
#pragma unroll
    for (int dt = 0; dt < 4; ++dt)
#pragma unroll
      for (int i = 0; i < 8; ++i) acc[dt][i] *= fv[i];

    // ---- P * V over this chunk ----
    for (int ks = 0; ks < CHUNK / 32; ++ks) {
      v16bf ap;
#pragma unroll
      for (int j = 0; j < 8; ++j) {
        ap[j]     = (bf16_t)Ssc[r * CHUNK + ks * 32 + half * 8 + j];
        ap[8 + j] = (bf16_t)Ssc[r * CHUNK + ks * 32 + 16 + half * 8 + j];
      }
#pragma unroll
      for (int dt = 0; dt < 4; ++dt) {
        const bf16_t* vp =
            vt + ((size_t)((b * NHEAD + h) * DHEAD + dt * 16 + r)) * SLEN +
            kc + ks * 32 + half * 8;
        v16bf bv = load_frag(vp);
        acc[dt] = wmma_bf16(ap, bv, acc[dt]);
      }
    }
  }

  // ---- normalize by row sums and write bf16 output [B,S,H] ----
  if (lane < 16) fac[lane] = 1.0f / l_r;
  float inv[8];
#pragma unroll
  for (int i = 0; i < 8; ++i) inv[i] = fac[i + half * 8];
#pragma unroll
  for (int dt = 0; dt < 4; ++dt)
#pragma unroll
    for (int i = 0; i < 8; ++i) {
      int qrow = q0 + i + half * 8;
      float v = acc[dt][i] * inv[i];
      ob[((size_t)(b * SLEN + qrow)) * HDIM + h * DHEAD + dt * 16 + r] =
          (bf16_t)v;
    }
}

// ---------------- LayerNorm (1 wave per row) ----------------
__global__ __launch_bounds__(32) void ln_kernel(
    const float* __restrict__ in, const float* __restrict__ g,
    const float* __restrict__ bta, float* __restrict__ outf,
    bf16_t* __restrict__ outb) {
  int row  = blockIdx.x;
  int lane = threadIdx.x;
  const float* rp = in + (size_t)row * HDIM;

  float vals[16];
  float s = 0.0f;
#pragma unroll
  for (int j = 0; j < 16; ++j) {
    float v = rp[lane + j * 32];
    vals[j] = v;
    s += v;
  }
#pragma unroll
  for (int off = 16; off > 0; off >>= 1) s += __shfl_xor(s, off, 32);
  float mean = s * (1.0f / HDIM);

  float s2 = 0.0f;
#pragma unroll
  for (int j = 0; j < 16; ++j) {
    float d = vals[j] - mean;
    s2 += d * d;
  }
#pragma unroll
  for (int off = 16; off > 0; off >>= 1) s2 += __shfl_xor(s2, off, 32);
  float rs = rsqrtf(s2 * (1.0f / HDIM) + 1e-5f);

#pragma unroll
  for (int j = 0; j < 16; ++j) {
    int c = lane + j * 32;
    float o = (vals[j] - mean) * rs * g[c] + bta[c];
    outf[(size_t)row * HDIM + c] = o;
    if (outb) outb[(size_t)row * HDIM + c] = (bf16_t)o;
  }
}

// ---------------- host orchestration ----------------
extern "C" void kernel_launch(void* const* d_in, const int* in_sizes, int n_in,
                              void* d_out, int out_size, void* d_ws, size_t ws_size,
                              hipStream_t stream) {
  (void)in_sizes; (void)n_in; (void)out_size; (void)ws_size;

  const float* x     = (const float*)d_in[0];
  const float* Wq    = (const float*)d_in[1];
  const float* Wk    = (const float*)d_in[2];
  const float* Wv    = (const float*)d_in[3];
  const float* Wo    = (const float*)d_in[4];
  const float* W1    = (const float*)d_in[5];
  const float* b1    = (const float*)d_in[6];
  const float* W2    = (const float*)d_in[7];
  const float* b2    = (const float*)d_in[8];
  const float* g_mha = (const float*)d_in[9];
  const float* bmha  = (const float*)d_in[10];
  const float* g_ff  = (const float*)d_in[11];
  const float* bff   = (const float*)d_in[12];
  float* out = (float*)d_out;

  // workspace carve-up
  const size_t WSZ   = (size_t)HDIM * HDIM * sizeof(bf16_t);   // 512 KB
  const size_t ACT_B = (size_t)MROWS * HDIM * sizeof(bf16_t);  // 8 MB
  const size_t ACT_F = (size_t)MROWS * HDIM * sizeof(float);   // 16 MB
  char* ws = (char*)d_ws;
  size_t off = 0;
  bf16_t* wt[6];
  for (int i = 0; i < 6; ++i) { wt[i] = (bf16_t*)(ws + off); off += WSZ; }
  bf16_t* xb   = (bf16_t*)(ws + off); off += ACT_B;
  bf16_t* qb   = (bf16_t*)(ws + off); off += ACT_B;
  bf16_t* kbuf = (bf16_t*)(ws + off); off += ACT_B;
  bf16_t* vb   = (bf16_t*)(ws + off); off += ACT_B;
  bf16_t* vt   = (bf16_t*)(ws + off); off += ACT_B;
  bf16_t* obuf = (bf16_t*)(ws + off); off += ACT_B;
  float*  y1   = (float*) (ws + off); off += ACT_F;   // also reused for FFN out
  float*  ln1  = (float*) (ws + off); off += ACT_F;
  bf16_t* ln1b = (bf16_t*)(ws + off); off += ACT_B;
  bf16_t* h1b  = (bf16_t*)(ws + off); off += ACT_B;

  const int NELEM = MROWS * HDIM;              // 4,194,304
  const int CVT_BLOCKS = NELEM / 256;          // 16384
  const int W_BLOCKS   = (HDIM * HDIM) / 256;  // 1024
  const int GEMM_BLOCKS = (MROWS / 128) * (HDIM / BN);     // 64 * 8 = 512
  const int ATTN_BLOCKS = BATCH * NHEAD * (SLEN / 16) / 2; // 2048

  // prep: weights -> bf16 transposed, x -> bf16
  const float* wsrc[6] = {Wq, Wk, Wv, Wo, W1, W2};
  for (int i = 0; i < 6; ++i)
    transpose_w_kernel<<<W_BLOCKS, 256, 0, stream>>>(wsrc[i], wt[i]);
  cvt_f32_bf16_kernel<<<CVT_BLOCKS, 256, 0, stream>>>(x, xb, NELEM);

  // QKV projections (bf16 out only)
  gemm_bf16_kernel<<<GEMM_BLOCKS, 256, 0, stream>>>(xb, wt[0], nullptr, nullptr,
                                                    nullptr, qb, MROWS, 0);
  gemm_bf16_kernel<<<GEMM_BLOCKS, 256, 0, stream>>>(xb, wt[1], nullptr, nullptr,
                                                    nullptr, kbuf, MROWS, 0);
  gemm_bf16_kernel<<<GEMM_BLOCKS, 256, 0, stream>>>(xb, wt[2], nullptr, nullptr,
                                                    nullptr, vb, MROWS, 0);

  // V -> [B,NH,DH,S] for contiguous B-fragments in P*V
  transpose_v_kernel<<<CVT_BLOCKS, 256, 0, stream>>>(vb, vt);

  // attention
  attn_kernel<<<ATTN_BLOCKS, 64, 0, stream>>>(qb, kbuf, vt, obuf);

  // out = x + o @ Wo   (fp32)
  gemm_bf16_kernel<<<GEMM_BLOCKS, 256, 0, stream>>>(obuf, wt[3], nullptr, x,
                                                    y1, nullptr, MROWS, 0);
  // ln_mha -> fp32 + bf16
  ln_kernel<<<MROWS, 32, 0, stream>>>(y1, g_mha, bmha, ln1, ln1b);

  // h1 = gelu(ln1 @ W1 + b1)  (bf16 out)
  gemm_bf16_kernel<<<GEMM_BLOCKS, 256, 0, stream>>>(ln1b, wt[4], b1, nullptr,
                                                    nullptr, h1b, MROWS, 1);
  // h = ln1 + (h1 @ W2 + b2)  (fp32, reuse y1)
  gemm_bf16_kernel<<<GEMM_BLOCKS, 256, 0, stream>>>(h1b, wt[5], b2, ln1,
                                                    y1, nullptr, MROWS, 0);
  // ln_ff -> final output
  ln_kernel<<<MROWS, 32, 0, stream>>>(y1, g_ff, bff, out, nullptr);
}
